// ODEGCN_C_17721035063791
// MI455X (gfx1250) — matrix operand order repaired and checked by reading, not compiled, not run on hardware
//
#include <hip/hip_runtime.h>
#include <stdint.h>

#define B_    8
#define N_    512
#define T_    24
#define F_    3
#define TOUT_ 12
#define C_    64
#define E_    8192

typedef __bf16 bf16_t;
typedef __attribute__((ext_vector_type(16))) __bf16 v16bf;
typedef __attribute__((ext_vector_type(8)))  float  v8f;

// ---------------------------------------------------------------------------
// Clean WMMA GEMM, B pre-transposed: C = A @ B (+Cacc)(+bias)(relu).
// A : M x K row-major (lda);  BT : N x K row-major (ldbt)  [i.e. B transposed]
// bf16 inputs / fp32 WMMA accumulation.
// PRECONDITIONS: M%16==0, K%32==0, (N/16)%NSUB==0, lda%4==0, ldbt%4==0.
// Per lane each fragment is two 8-float runs, 16 apart -> 4x b128 each.
// No lane-divergent guards; only wave-uniform work-range exit (EXEC rule ok).
// One wave -> 16 x (16*NSUB) strip; 4 waves/block; grid.z = batch.
// ---------------------------------------------------------------------------
template <int NSUB>
__global__ void __launch_bounds__(128)
wmma_gemm_fast(const float* __restrict__ A, long long sAz, int lda,
               const float* __restrict__ BT, long long sBz, int ldbt,
               const float* __restrict__ bias,
               const float* Cacc,
               float* Cout, long long sCz, int ldc,
               int M, int Nn, int K, int relu)
{
  const int wave = threadIdx.x >> 5;
  const int lane = threadIdx.x & 31;
  const int tilesM  = M >> 4;
  const int groupsN = (Nn >> 4) / NSUB;
  const long long work = (long long)blockIdx.x * 4 + wave;
  if (work >= (long long)tilesM * groupsN) return;    // wave-uniform exit
  const int tm = (int)(work / groupsN);
  const int tg = (int)(work % groupsN);
  const int z  = blockIdx.z;

  const int rowL = lane & 15;
  const int half = lane >> 4;
  const int row  = tm * 16 + rowL;
  const int n0   = tg * (NSUB * 16) + rowL;

  // A fragment: elems i<8 -> K=k0+half*8+i ; i>=8 -> +16  (two b128 pairs)
  const float* arow = A + (long long)z * sAz + (long long)row * lda + (half << 3);
  // B fragment: elems i<8 -> K=k0+half*8+i ; i>=8 -> +16  (two b128 pairs)
  const float* bt[NSUB];
#pragma unroll
  for (int j = 0; j < NSUB; ++j)
    bt[j] = BT + (long long)z * sBz + (long long)(n0 + j * 16) * ldbt + (half << 3);

  v8f zero = {};
  v8f acc[NSUB];
#pragma unroll
  for (int j = 0; j < NSUB; ++j) acc[j] = zero;

  for (int k0 = 0; k0 < K; k0 += 32) {
    v16bf a;
    {
      const float4 a0 = *(const float4*)(arow + k0);
      const float4 a1 = *(const float4*)(arow + k0 + 4);
      const float4 a2 = *(const float4*)(arow + k0 + 16);
      const float4 a3 = *(const float4*)(arow + k0 + 20);
      a[0]=(bf16_t)a0.x;  a[1]=(bf16_t)a0.y;  a[2]=(bf16_t)a0.z;  a[3]=(bf16_t)a0.w;
      a[4]=(bf16_t)a1.x;  a[5]=(bf16_t)a1.y;  a[6]=(bf16_t)a1.z;  a[7]=(bf16_t)a1.w;
      a[8]=(bf16_t)a2.x;  a[9]=(bf16_t)a2.y;  a[10]=(bf16_t)a2.z; a[11]=(bf16_t)a2.w;
      a[12]=(bf16_t)a3.x; a[13]=(bf16_t)a3.y; a[14]=(bf16_t)a3.z; a[15]=(bf16_t)a3.w;
    }
#pragma unroll
    for (int j = 0; j < NSUB; ++j) {
      const float4 b0 = *(const float4*)(bt[j] + k0);
      const float4 b1 = *(const float4*)(bt[j] + k0 + 4);
      const float4 b2 = *(const float4*)(bt[j] + k0 + 16);
      const float4 b3 = *(const float4*)(bt[j] + k0 + 20);
      v16bf b;
      b[0]=(bf16_t)b0.x;  b[1]=(bf16_t)b0.y;  b[2]=(bf16_t)b0.z;  b[3]=(bf16_t)b0.w;
      b[4]=(bf16_t)b1.x;  b[5]=(bf16_t)b1.y;  b[6]=(bf16_t)b1.z;  b[7]=(bf16_t)b1.w;
      b[8]=(bf16_t)b2.x;  b[9]=(bf16_t)b2.y;  b[10]=(bf16_t)b2.z; b[11]=(bf16_t)b2.w;
      b[12]=(bf16_t)b3.x; b[13]=(bf16_t)b3.y; b[14]=(bf16_t)b3.z; b[15]=(bf16_t)b3.w;
      acc[j] = __builtin_amdgcn_wmma_f32_16x16x32_bf16(false, a, false, b,
                                                       (short)0, acc[j], false, false);
    }
  }

  float* Cw = Cout + (long long)z * sCz;
  const float* Cr = Cacc ? (Cacc + (long long)z * sCz) : nullptr;
#pragma unroll
  for (int j = 0; j < NSUB; ++j) {
    const int n = n0 + j * 16;
    const float bv = bias ? bias[n] : 0.f;
#pragma unroll
    for (int r = 0; r < 8; ++r) {
      const int m = tm * 16 + r + (half << 3);
      float v = acc[j][r] + bv;
      if (Cr)   v += Cr[(long long)m * ldc + n];
      if (relu) v = v > 0.f ? v : 0.f;
      Cw[(long long)m * ldc + n] = v;
    }
  }
}

// Tiled LDS transpose: dst[c][r] = src[r][c]; src rows x cols, batched via z.
__global__ void transpose_kernel(const float* __restrict__ src, long long sSz,
                                 int rows, int cols,
                                 float* __restrict__ dst, long long sDz)
{
  __shared__ float tile[32][33];
  const int z  = blockIdx.z;
  const int c0 = blockIdx.x * 32;
  const int r0 = blockIdx.y * 32;
  const float* S = src + (long long)z * sSz;
  float* D = dst + (long long)z * sDz;
#pragma unroll
  for (int rr = threadIdx.y; rr < 32; rr += 8) {
    const int r = r0 + rr, c = c0 + threadIdx.x;
    if (r < rows && c < cols) tile[rr][threadIdx.x] = S[(long long)r * cols + c];
  }
  __syncthreads();
#pragma unroll
  for (int rr = threadIdx.y; rr < 32; rr += 8) {
    const int n = c0 + rr, k = r0 + threadIdx.x;
    if (n < cols && k < rows) D[(long long)n * rows + k] = tile[threadIdx.x][rr];
  }
}

// First-block MLP1: out[row,n] = relu(bias[n] + sum_{k<3} x[row*3+k]*W[k*128+n])
// K=3 => 3/32 WMMA K-utilization; memory-bound, VALU is the right tool.
__global__ void mlp1_k3_kernel(const float* __restrict__ x,
                               const float* __restrict__ W,   // 3 x 128
                               const float* __restrict__ bias,
                               float* __restrict__ out, int Mrows)
{
  const long long idx = (long long)blockIdx.x * blockDim.x + threadIdx.x;
  if (idx >= (long long)Mrows * 128) return;
  const int n   = (int)(idx & 127);
  const long long row = idx >> 7;
  const float x0 = x[row * 3 + 0], x1 = x[row * 3 + 1], x2 = x[row * 3 + 2];
  float v = bias[n] + x0 * W[n] + x1 * W[128 + n] + x2 * W[256 + n];
  out[idx] = v > 0.f ? v : 0.f;
}

// pred2: out[m,n] = bias[n] + sum_k A[m*96+k] * W[k*12+n]   (4096 x 12 x 96)
__global__ void pred2_kernel(const float* __restrict__ A,
                             const float* __restrict__ W,   // 96 x 12
                             const float* __restrict__ bias,
                             float* __restrict__ out, int Mrows)
{
  const long long idx = (long long)blockIdx.x * blockDim.x + threadIdx.x;
  if (idx >= (long long)Mrows * TOUT_) return;
  const int n = (int)(idx % TOUT_);
  const long long m = idx / TOUT_;
  float v = bias[n];
  const float* ar = A + m * 96;
#pragma unroll
  for (int k = 0; k < 96; ++k) v += ar[k] * W[k * TOUT_ + n];
  out[idx] = v;
}

// W_eff = (W .* clip(d,0,1)) @ W^T (symmetric!) and W2_eff likewise (24x24)
__global__ void eff_kernel(const float* __restrict__ w,  const float* __restrict__ d,
                           const float* __restrict__ w2, const float* __restrict__ d2,
                           float* __restrict__ weff, float* __restrict__ w2eff)
{
  const int t = blockIdx.x * blockDim.x + threadIdx.x;
  if (t < C_ * C_) {
    const int i = t / C_, m = t % C_;
    float s = 0.f;
    for (int c = 0; c < C_; ++c) {
      float dc = fminf(fmaxf(d[c], 0.f), 1.f);
      s += w[i * C_ + c] * dc * w[m * C_ + c];
    }
    weff[t] = s;
  }
  if (t < T_ * T_) {
    const int i = t / T_, m = t % T_;
    float s = 0.f;
    for (int c = 0; c < T_; ++c) {
      float dc = fminf(fmaxf(d2[c], 0.f), 1.f);
      s += w2[i * T_ + c] * dc * w2[m * T_ + c];
    }
    w2eff[t] = s;
  }
}

// z = relu(3*sigmoid(alpha)*xa + 6*hw + 6*hw2 - 11*m)   (one thread per (b,n,c))
__global__ void ode_fuse_kernel(const float* __restrict__ m,
                                const float* __restrict__ xa,
                                const float* __restrict__ hw,
                                const float* __restrict__ w2eff,
                                const float* __restrict__ alpha,
                                float* __restrict__ z)
{
  const long long idx = (long long)blockIdx.x * blockDim.x + threadIdx.x;
  const long long total = (long long)B_ * N_ * C_;
  if (idx >= total) return;
  const int c = (int)(idx % C_);
  const int n = (int)((idx / C_) % N_);
  const int b = (int)(idx / ((long long)C_ * N_));
  const long long base = (((long long)b * N_ + n) * T_) * C_ + c;
  const float a = 1.f / (1.f + expf(-alpha[n]));
  float mreg[T_];
#pragma unroll
  for (int t = 0; t < T_; ++t) mreg[t] = m[base + (long long)t * C_];
#pragma unroll 1
  for (int t = 0; t < T_; ++t) {
    float h2 = 0.f;
#pragma unroll
    for (int tp = 0; tp < T_; ++tp) h2 += mreg[tp] * w2eff[tp * T_ + t];
    const long long o = base + (long long)t * C_;
    float v = 3.f * a * xa[o] + 6.f * hw[o] + 6.f * h2 - 11.f * mreg[t];
    z[o] = v > 0.f ? v : 0.f;
  }
}

// In-place per-node BatchNorm over (B,T,C) for each node n (one block per n)
__global__ void bn_kernel(float* __restrict__ h, const float* __restrict__ gamma,
                          const float* __restrict__ beta)
{
  const int n  = blockIdx.x;
  const int tc = T_ * C_;
  const int cnt = B_ * tc;
  __shared__ float ssum[256], ssq[256];
  float s = 0.f, q = 0.f;
  for (int i = threadIdx.x; i < cnt; i += blockDim.x) {
    const int b = i / tc, r = i % tc;
    const float v = h[(((long long)b * N_ + n) * tc) + r];
    s += v; q += v * v;
  }
  ssum[threadIdx.x] = s; ssq[threadIdx.x] = q;
  __syncthreads();
  for (int off = 128; off > 0; off >>= 1) {
    if (threadIdx.x < (unsigned)off) {
      ssum[threadIdx.x] += ssum[threadIdx.x + off];
      ssq[threadIdx.x]  += ssq[threadIdx.x + off];
    }
    __syncthreads();
  }
  const float mean = ssum[0] / (float)cnt;
  const float var  = ssq[0] / (float)cnt - mean * mean;
  const float sc   = gamma[n] * rsqrtf(var + 1e-5f);
  const float sh   = beta[n] - mean * sc;
  for (int i = threadIdx.x; i < cnt; i += blockDim.x) {
    const int b = i / tc, r = i % tc;
    const long long o = (((long long)b * N_ + n) * tc) + r;
    h[o] = h[o] * sc + sh;
  }
}

__global__ void fill_kernel(float* __restrict__ p, long long n, float v) {
  const long long i = (long long)blockIdx.x * blockDim.x + threadIdx.x;
  if (i < n) p[i] = v;
}

// agg[b,dst,t,c] = max over in-edges of pooled[b,src,t,c]; pooled >= 0 so int
// atomicMax on the bit pattern is order-preserving; 0-init reproduces -inf->0.
__global__ void scatter_max_kernel(const float* __restrict__ pooled,
                                   const int* __restrict__ src,
                                   const int* __restrict__ dst,
                                   float* __restrict__ agg)
{
  const long long idx = (long long)blockIdx.x * blockDim.x + threadIdx.x;
  const int tc = T_ * C_;
  if (idx >= (long long)E_ * tc) return;
  const int r = (int)(idx % tc);
  const int e = (int)(idx / tc);
  const int s = src[e], d = dst[e];
#pragma unroll
  for (int b = 0; b < B_; ++b) {
    const float v = pooled[(((long long)b * N_ + s) * tc) + r];
    atomicMax((int*)(agg + (((long long)b * N_ + d) * tc) + r), __float_as_int(v));
  }
}

__global__ void combine_max_kernel(const float* __restrict__ src,
                                   float* __restrict__ dst, int first, long long n)
{
  const long long i = (long long)blockIdx.x * blockDim.x + threadIdx.x;
  if (i >= n) return;
  dst[i] = first ? src[i] : fmaxf(dst[i], src[i]);
}

// ---------------------------------------------------------------------------
extern "C" void kernel_launch(void* const* d_in, const int* in_sizes, int n_in,
                              void* d_out, int out_size, void* d_ws, size_t ws_size,
                              hipStream_t stream)
{
  (void)in_sizes; (void)n_in; (void)out_size; (void)ws_size;
  const float* x     = (const float*)d_in[0];
  const float* A_hat = (const float*)d_in[1];
  const int*   esrc  = (const int*)d_in[2];
  const int*   edst  = (const int*)d_in[3];
  auto P = [&](int i) { return (const float*)d_in[i]; };

  const long long Pn  = (long long)B_ * N_ * T_ * C_;   // 6.29M floats
  const long long PnB = (long long)N_ * T_ * C_;        // per-batch stride
  float* ws     = (float*)d_ws;
  float* H      = ws;                 // (B,N,T,C)  block activations / z / BN out
  float* MID    = ws + Pn;            // 2*Pn: MLP hidden, then [xa | mT->hw], then [pooled | agg]
  float* MBUF   = ws + 3 * Pn;        // m / sage out
  float* MAXB   = ws + 4 * Pn;        // running max over branches
  float* PMID   = ws + 5 * Pn;        // (B*N, 96)
  float* WEFF   = PMID + (long long)B_ * N_ * 96;
  float* W2EFF  = WEFF + C_ * C_;
  float* M1WT   = W2EFF + T_ * T_;    // 128 x 64
  float* M2WT   = M1WT + 128 * C_;    // 64 x 128
  float* POOLT  = M2WT + C_ * 128;    // 64 x 64
  float* SELFT  = POOLT + C_ * C_;
  float* NEIGHT = SELFT + C_ * C_;
  float* PRED1T = NEIGHT + C_ * C_;   // 96 x 1536

  const int MROWS = B_ * N_ * T_;     // 98304
  const int TC    = T_ * C_;          // 1536

  auto gemm = [&](const float* A, long long sAz, int lda,
                  const float* BT, long long sBz, int ldbt,
                  const float* bias, const float* Cacc,
                  float* Cout, long long sCz, int ldc,
                  int M, int Nn, int K, int relu, int batches) {
    const int tilesM = M >> 4;
    const int tilesN = Nn >> 4;
    const int nsub   = (tilesN % 4 == 0) ? 4 : ((tilesN % 2 == 0) ? 2 : 1);
    const long long work = (long long)tilesM * (tilesN / nsub);
    dim3 g((unsigned)((work + 3) / 4), 1, (unsigned)batches);
    if (nsub == 4)
      wmma_gemm_fast<4><<<g, dim3(128), 0, stream>>>(A, sAz, lda, BT, sBz, ldbt,
          bias, Cacc, Cout, sCz, ldc, M, Nn, K, relu);
    else if (nsub == 2)
      wmma_gemm_fast<2><<<g, dim3(128), 0, stream>>>(A, sAz, lda, BT, sBz, ldbt,
          bias, Cacc, Cout, sCz, ldc, M, Nn, K, relu);
    else
      wmma_gemm_fast<1><<<g, dim3(128), 0, stream>>>(A, sAz, lda, BT, sBz, ldbt,
          bias, Cacc, Cout, sCz, ldc, M, Nn, K, relu);
  };
  auto transpose = [&](const float* src, long long sSz, int rows, int cols,
                       float* dst, long long sDz, int batches) {
    dim3 g((unsigned)((cols + 31) / 32), (unsigned)((rows + 31) / 32), (unsigned)batches);
    transpose_kernel<<<g, dim3(32, 8), 0, stream>>>(src, sSz, rows, cols, dst, sDz);
  };

  const int sp = 4 + 66;  // params: 3 branches * 2 blocks * 11 leaves, then sage/pred

  // shared (branch-invariant) weight transposes
  transpose(P(sp),     0, C_, C_, POOLT,  0, 1);   // pool.w   64x64 -> 64x64
  transpose(P(sp + 2), 0, C_, C_, SELFT,  0, 1);   // self.w
  transpose(P(sp + 4), 0, C_, C_, NEIGHT, 0, 1);   // neigh.w
  transpose(P(sp + 5), 0, TC, 96, PRED1T, 0, 1);   // pred1.w  1536x96 -> 96x1536

  for (int br = 0; br < 3; ++br) {
    for (int bl = 0; bl < 2; ++bl) {
      const int p = 4 + (br * 2 + bl) * 11;
      const float *m1w = P(p),     *m1b = P(p + 1);
      const float *m2w = P(p + 2), *m2b = P(p + 3);
      const float *alp = P(p + 4);
      const float *ow  = P(p + 5), *od  = P(p + 6);
      const float *ow2 = P(p + 7), *od2 = P(p + 8);
      const float *gam = P(p + 9), *bet = P(p + 10);

      eff_kernel<<<dim3(16), dim3(256), 0, stream>>>(ow, od, ow2, od2, WEFF, W2EFF);

      // MLP1: relu(A @ W1 + b1) -> MID (ld 128)
      if (bl == 0) {
        const long long tot = (long long)MROWS * 128;
        mlp1_k3_kernel<<<dim3((unsigned)((tot + 255) / 256)), 256, 0, stream>>>(
            x, m1w, m1b, MID, MROWS);
      } else {
        transpose(m1w, 0, C_, 128, M1WT, 0, 1);          // 64x128 -> 128x64
        gemm(H, 0, C_, M1WT, 0, C_, m1b, nullptr, MID, 0, 128, MROWS, 128, C_, 1, 1);
      }
      // MLP2: MID @ W2 + b2 -> m (MBUF)
      transpose(m2w, 0, 128, C_, M2WT, 0, 1);            // 128x64 -> 64x128
      gemm(MID, 0, 128, M2WT, 0, 128, m2b, nullptr, MBUF, 0, C_, MROWS, C_, 128, 0, 1);
      // m^T per batch (512x1536 -> 1536x512) into MID[Pn:2Pn]
      transpose(MBUF, PnB, N_, TC, MID + Pn, PnB, B_);
      // graph conv: per batch (A_hat 512x512) @ m_b -> xa (MID[0:Pn])
      gemm(A_hat, 0, N_, MID + Pn, PnB, N_, nullptr, nullptr,
           MID, PnB, TC, N_, TC, N_, 0, B_);
      // hw = m @ W_eff -> MID[Pn:2Pn] (W_eff symmetric => is its own transpose)
      gemm(MBUF, 0, C_, WEFF, 0, C_, nullptr, nullptr, MID + Pn, 0, C_, MROWS, C_, C_, 0, 1);
      // fused Euler step + relu -> H
      const long long tot = (long long)B_ * N_ * C_;
      ode_fuse_kernel<<<dim3((unsigned)((tot + 255) / 256)), 256, 0, stream>>>(
          MBUF, MID, MID + Pn, W2EFF, alp, H);
      // BatchNorm (training-mode batch stats) in place on H
      bn_kernel<<<dim3(N_), 256, 0, stream>>>(H, gam, bet);
    }

    // SAGE 'pool' aggregator (shared params)
    const float *pb = P(sp + 1), *sb = P(sp + 3);
    // pooled = relu(H @ Wpool + b) -> MID[0:Pn]
    gemm(H, 0, C_, POOLT, 0, C_, pb, nullptr, MID, 0, C_, MROWS, C_, C_, 1, 1);
    // agg = 0 -> MID[Pn:2Pn]
    fill_kernel<<<dim3((unsigned)((Pn + 255) / 256)), 256, 0, stream>>>(MID + Pn, Pn, 0.f);
    const long long sidx = (long long)E_ * TC;
    scatter_max_kernel<<<dim3((unsigned)((sidx + 255) / 256)), 256, 0, stream>>>(
        MID, esrc, edst, MID + Pn);
    // out = H @ Wself + bself  (+ agg @ Wneigh accumulated) -> MBUF
    gemm(H, 0, C_, SELFT, 0, C_, sb, nullptr, MBUF, 0, C_, MROWS, C_, C_, 0, 1);
    gemm(MID + Pn, 0, C_, NEIGHT, 0, C_, nullptr, MBUF, MBUF, 0, C_, MROWS, C_, C_, 0, 1);
    // running max across branches
    combine_max_kernel<<<dim3((unsigned)((Pn + 255) / 256)), 256, 0, stream>>>(
        MBUF, MAXB, br == 0 ? 1 : 0, Pn);
  }

  // predictor: relu((B*N,1536) @ (1536,96) + b) -> PMID,  then (96->12) -> d_out
  gemm(MAXB, 0, TC, PRED1T, 0, TC, P(sp + 6), nullptr, PMID, 0, 96,
       B_ * N_, 96, TC, 1, 1);
  {
    const long long tot = (long long)B_ * N_ * TOUT_;
    pred2_kernel<<<dim3((unsigned)((tot + 255) / 256)), 256, 0, stream>>>(
        PMID, P(sp + 7), P(sp + 8), (float*)d_out, B_ * N_);
  }
}
